// MultiHeadSelfAttention_41987600285882
// MI455X (gfx1250) — matrix operand order repaired and checked by reading, not compiled, not run on hardware
//
#include <hip/hip_runtime.h>
#include <stdint.h>

// ---------------- CDNA5 WMMA types ----------------
typedef __attribute__((ext_vector_type(16))) __bf16    v16bf;
typedef __attribute__((ext_vector_type(8)))  float     v8f;
typedef __attribute__((ext_vector_type(4)))  unsigned int u32x4;

struct Frag256 { u32x4 lo; u32x4 hi; };

static __device__ __forceinline__ v16bf make_frag(const unsigned short* p_lo,
                                                  const unsigned short* p_hi) {
  Frag256 f;
  f.lo = *(const u32x4*)p_lo;
  f.hi = *(const u32x4*)p_hi;
  return __builtin_bit_cast(v16bf, f);
}

static __device__ __forceinline__ v8f wmma_bf16(v16bf a, v16bf b, v8f c) {
  // D = A(16x32 bf16) * B(32x16 bf16) + C(16x16 f32)
  return __builtin_amdgcn_wmma_f32_16x16x32_bf16(false, a, false, b,
                                                 (short)0, c, false, false);
}

static __device__ __forceinline__ unsigned short f32_to_bf16(float f) {
  unsigned int u = __float_as_uint(f);
  unsigned int r = u + 0x7FFFu + ((u >> 16) & 1u);   // round-to-nearest-even
  return (unsigned short)(r >> 16);
}

// CDNA5 async global->LDS copy, 16B per lane, tracked by ASYNCcnt.
// lds operand = workgroup-relative LDS address (low 32 bits of generic ptr);
// hardware adds LDS_BASE (see ISA 15.18.3).
static __device__ __forceinline__ void async_b128(const void* gptr, void* lptr) {
  unsigned int lds = (unsigned int)(size_t)lptr;
  unsigned long long ga = (unsigned long long)(size_t)gptr;
  asm volatile("global_load_async_to_lds_b128 %0, %1, off"
               :: "v"(lds), "v"(ga) : "memory");
}
static __device__ __forceinline__ void wait_async0() {
  asm volatile("s_wait_asynccnt 0x0" ::: "memory");
}

#define NEGF (-3.402823466e38f)   // matches jnp.finfo(float32).min semantics
#define SCALE 0.125f              // 1/sqrt(64)

// ---------------- fp32 -> bf16 convert ----------------
__global__ void cvt_f32_bf16(const float* __restrict__ src,
                             unsigned short* __restrict__ dst, int n) {
  int i = blockIdx.x * blockDim.x + threadIdx.x;
  if (i < n) dst[i] = f32_to_bf16(src[i]);
}

// ---------------- bf16 GEMM: C[M,N] = A[M,K] * B[K,N] ----------------
// mode 0: scatter to Q/K/V bf16 [B=4,H=16,S=2048,HD=64]  (N = 3072)
// mode 1: write fp32 row-major Cf[M,N]
#define BM 128
#define BN 128
#define BK 32
#define LDA 40
#define LDB 40

__global__ __launch_bounds__(256)
void gemm_bf16(const unsigned short* __restrict__ A,
               const unsigned short* __restrict__ Bm,
               int K, int N, int mode,
               unsigned short* __restrict__ Qw,
               unsigned short* __restrict__ Kw,
               unsigned short* __restrict__ Vw,
               float* __restrict__ Cf) {
  __shared__ __align__(16) unsigned short As[BM * LDA];
  __shared__ __align__(16) unsigned short Bs[BN * LDB];

  const int t    = threadIdx.x;
  const int lane = t & 31;
  const int wave = t >> 5;          // 0..7
  const int wm   = wave & 3;        // M quadrant (32 rows each)
  const int wn   = wave >> 2;       // N half (64 cols each)
  const int m0   = blockIdx.x * BM;
  const int n0   = blockIdx.y * BN;
  const int half = lane >> 4;
  const int l16  = lane & 15;

  v8f acc[2][4];
#pragma unroll
  for (int i = 0; i < 2; ++i)
#pragma unroll
    for (int j = 0; j < 4; ++j) acc[i][j] = (v8f)0.0f;

  for (int k0 = 0; k0 < K; k0 += BK) {
    __syncthreads();
    // stage A tile (128 rows x 32 halves) via async global->LDS (no VGPR trip)
#pragma unroll
    for (int c = t; c < 512; c += 256) {
      int r = c >> 2, cb = c & 3;
      async_b128(A + (size_t)(m0 + r) * K + k0 + cb * 8,
                 As + r * LDA + cb * 8);
    }
    // stage B tile transposed: Bs[n][k] (2-byte transpose, manual path)
    {
      int r = t >> 3, nb = t & 7;
      const unsigned short* src = Bm + (size_t)(k0 + r) * N + n0 + nb * 16;
      u32x4 v0 = *(const u32x4*)src;
      u32x4 v1 = *(const u32x4*)(src + 8);
#pragma unroll
      for (int i = 0; i < 4; ++i) {
        unsigned int w0 = v0[i], w1 = v1[i];
        Bs[(nb * 16 + 2 * i + 0) * LDB + r] = (unsigned short)(w0 & 0xffff);
        Bs[(nb * 16 + 2 * i + 1) * LDB + r] = (unsigned short)(w0 >> 16);
        Bs[(nb * 16 + 8 + 2 * i + 0) * LDB + r] = (unsigned short)(w1 & 0xffff);
        Bs[(nb * 16 + 8 + 2 * i + 1) * LDB + r] = (unsigned short)(w1 >> 16);
      }
    }
    wait_async0();
    __syncthreads();

    v16bf afr[2];
#pragma unroll
    for (int i = 0; i < 2; ++i) {
      int m = wm * 32 + i * 16 + l16;
      afr[i] = make_frag(As + m * LDA + 8 * half, As + m * LDA + 16 + 8 * half);
    }
#pragma unroll
    for (int jn = 0; jn < 4; ++jn) {
      int n = wn * 64 + jn * 16 + l16;
      v16bf bfr = make_frag(Bs + n * LDB + 8 * half, Bs + n * LDB + 16 + 8 * half);
#pragma unroll
      for (int i = 0; i < 2; ++i) acc[i][jn] = wmma_bf16(afr[i], bfr, acc[i][jn]);
    }
  }

  // epilogue: D layout row = j + 8*(lane/16), col = lane%16
#pragma unroll
  for (int i = 0; i < 2; ++i)
#pragma unroll
    for (int jn = 0; jn < 4; ++jn) {
      int gn = n0 + wn * 64 + jn * 16 + l16;
#pragma unroll
      for (int j = 0; j < 8; ++j) {
        int gm = m0 + wm * 32 + i * 16 + j + 8 * half;
        float v = acc[i][jn][j];
        if (mode == 1) {
          Cf[(size_t)gm * N + gn] = v;
        } else {
          int mat = gn >> 10, rem = gn & 1023;
          int h = rem >> 6, hd = rem & 63;
          int b = gm >> 11, s = gm & 2047;
          unsigned short* dst = (mat == 0) ? Qw : ((mat == 1) ? Kw : Vw);
          dst[(((size_t)(b * 16 + h)) * 2048 + s) * 64 + hd] = f32_to_bf16(v);
        }
      }
    }
}

// ---------------- flash attention (causal + pad mask, online softmax) ------
// grid: (S/64, B*H); block 128 threads = 4 waves, each wave owns 16 q rows.
__global__ __launch_bounds__(128)
void attn_kernel(const unsigned short* __restrict__ Qw,
                 const unsigned short* __restrict__ Kw,
                 const unsigned short* __restrict__ Vw,
                 const int* __restrict__ amask,
                 unsigned short* __restrict__ attnO) {
  __shared__ __align__(16) unsigned short Ks[32 * 72];   // [key][hd]
  __shared__ __align__(16) unsigned short Vs[64 * 40];   // [hd][key] (transposed)
  __shared__ __align__(16) unsigned short Ps[4][16 * 40];// per-wave P scratch

  const int t    = threadIdx.x;
  const int lane = t & 31;
  const int wv   = t >> 5;
  const int half = lane >> 4;
  const int l16  = lane & 15;
  const int bh   = blockIdx.y;           // b*16 + h
  const int b    = bh >> 4, h = bh & 15;
  const int q0b  = blockIdx.x * 64;
  const int q0w  = q0b + wv * 16;
  const size_t baseBH = (size_t)bh * 2048 * 64;

  // Q fragments (A layout), kept in registers for the whole KV sweep
  v16bf aq[2];
  {
    const unsigned short* qrow = Qw + baseBH + (size_t)(q0w + l16) * 64;
    aq[0] = make_frag(qrow + 8 * half,      qrow + 16 + 8 * half);
    aq[1] = make_frag(qrow + 32 + 8 * half, qrow + 48 + 8 * half);
  }

  v8f o[4];
#pragma unroll
  for (int c = 0; c < 4; ++c) o[c] = (v8f)0.0f;
  float mst[8], lst[8];
#pragma unroll
  for (int j = 0; j < 8; ++j) { mst[j] = NEGF; lst[j] = 0.0f; }

  const int nch = (q0b + 64 + 31) >> 5;  // 32-key chunks up to causal limit
  for (int ch = 0; ch < nch; ++ch) {
    const int k0 = ch * 32;
    __syncthreads();
    // cooperative staging: K row-major via async copies, V transposed manually
    {
      int r = t >> 2, cb = t & 3;
      const unsigned short* sk = Kw + baseBH + (size_t)(k0 + r) * 64 + cb * 16;
      async_b128(sk,     Ks + r * 72 + cb * 16);
      async_b128(sk + 8, Ks + r * 72 + cb * 16 + 8);
      const unsigned short* sv = Vw + baseBH + (size_t)(k0 + r) * 64 + cb * 16;
      u32x4 v0 = *(const u32x4*)sv;
      u32x4 v1 = *(const u32x4*)(sv + 8);
#pragma unroll
      for (int i = 0; i < 4; ++i) {
        unsigned int w0 = v0[i], w1 = v1[i];
        Vs[(cb * 16 + 2 * i + 0) * 40 + r] = (unsigned short)(w0 & 0xffff);
        Vs[(cb * 16 + 2 * i + 1) * 40 + r] = (unsigned short)(w0 >> 16);
        Vs[(cb * 16 + 8 + 2 * i + 0) * 40 + r] = (unsigned short)(w1 & 0xffff);
        Vs[(cb * 16 + 8 + 2 * i + 1) * 40 + r] = (unsigned short)(w1 >> 16);
      }
    }
    wait_async0();
    __syncthreads();
    if (k0 > q0w + 15) continue;   // wave-uniform skip; barriers stay aligned

    // scores S[16q][32keys] = Q * K^T  (two 16-key tiles, HD contracted 32+32)
    v8f S[2];
    S[0] = (v8f)0.0f; S[1] = (v8f)0.0f;
#pragma unroll
    for (int tt = 0; tt < 2; ++tt) {
      const unsigned short* kr = Ks + (tt * 16 + l16) * 72;
      v16bf b0 = make_frag(kr + 8 * half,      kr + 16 + 8 * half);
      v16bf b1 = make_frag(kr + 32 + 8 * half, kr + 48 + 8 * half);
      S[tt] = wmma_bf16(aq[0], b0, S[tt]);
      S[tt] = wmma_bf16(aq[1], b1, S[tt]);
    }

    const int key0 = k0 + l16, key1 = k0 + 16 + l16;
    const int am0 = amask[b * 2048 + key0];
    const int am1 = amask[b * 2048 + key1];

    float s0[8], s1[8];
#pragma unroll
    for (int j = 0; j < 8; ++j) {
      int q = q0w + j + 8 * half;
      float v0 = S[0][j] * SCALE; if (key0 > q || am0 == 0) v0 = NEGF;
      float v1 = S[1][j] * SCALE; if (key1 > q || am1 == 0) v1 = NEGF;
      s0[j] = v0; s1[j] = v1;
    }

    // online softmax update (reductions stay within each 16-lane half)
#pragma unroll
    for (int j = 0; j < 8; ++j) {
      float mx = fmaxf(s0[j], s1[j]);
#pragma unroll
      for (int d = 1; d < 16; d <<= 1) mx = fmaxf(mx, __shfl_xor(mx, d, 32));
      float mn = fmaxf(mst[j], mx);
      float sc = __expf(mst[j] - mn);
      float p0 = __expf(s0[j] - mn);
      float p1 = __expf(s1[j] - mn);
      float rs = p0 + p1;
#pragma unroll
      for (int d = 1; d < 16; d <<= 1) rs += __shfl_xor(rs, d, 32);
      lst[j] = lst[j] * sc + rs;
      mst[j] = mn;
#pragma unroll
      for (int c = 0; c < 4; ++c) o[c][j] *= sc;
      // P in D layout -> per-wave LDS (same-wave DS ops are in-order)
      Ps[wv][(j + 8 * half) * 40 + l16]      = f32_to_bf16(p0);
      Ps[wv][(j + 8 * half) * 40 + 16 + l16] = f32_to_bf16(p1);
    }

    // O += P(16x32) * V(32x64)
    {
      const unsigned short* pr = Ps[wv] + l16 * 40;
      v16bf pa = make_frag(pr + 8 * half, pr + 16 + 8 * half);
#pragma unroll
      for (int c = 0; c < 4; ++c) {
        const unsigned short* vr = Vs + (c * 16 + l16) * 40;
        v16bf bv = make_frag(vr + 8 * half, vr + 16 + 8 * half);
        o[c] = wmma_bf16(pa, bv, o[c]);
      }
    }
  }

  // normalize and merge heads: attnO[b*2048+q][h*64 + hd] (bf16)
#pragma unroll
  for (int j = 0; j < 8; ++j) {
    float inv = 1.0f / lst[j];
    int q = q0w + j + 8 * half;
    size_t rowo = ((size_t)(b * 2048 + q)) * 1024 + h * 64;
#pragma unroll
    for (int c = 0; c < 4; ++c)
      attnO[rowo + c * 16 + l16] = f32_to_bf16(o[c][j] * inv);
  }
}

// ---------------- host launch ----------------
extern "C" void kernel_launch(void* const* d_in, const int* in_sizes, int n_in,
                              void* d_out, int out_size, void* d_ws, size_t ws_size,
                              hipStream_t stream) {
  const float* x     = (const float*)d_in[0];   // [4,2048,1024]
  const int*   am    = (const int*)  d_in[1];   // [4,2048]
  const float* Wqkv  = (const float*)d_in[2];   // [1024,3072]
  const float* Wout  = (const float*)d_in[3];   // [1024,1024]
  float*       out   = (float*)d_out;           // [4,2048,1024]

  // workspace layout (bf16 halves), ~92 MB total
  unsigned short* xb    = (unsigned short*)d_ws;          // 8388608
  unsigned short* wqkvb = xb    + 8388608;                // 3145728
  unsigned short* woutb = wqkvb + 3145728;                // 1048576
  unsigned short* Qw    = woutb + 1048576;                // 8388608
  unsigned short* Kw    = Qw    + 8388608;
  unsigned short* Vw    = Kw    + 8388608;
  unsigned short* attnb = Vw    + 8388608;                // 8388608

  cvt_f32_bf16<<<(8388608 + 255) / 256, 256, 0, stream>>>(x,    xb,    8388608);
  cvt_f32_bf16<<<(3145728 + 255) / 256, 256, 0, stream>>>(Wqkv, wqkvb, 3145728);
  cvt_f32_bf16<<<(1048576 + 255) / 256, 256, 0, stream>>>(Wout, woutb, 1048576);

  dim3 gQKV(8192 / BM, 3072 / BN);
  gemm_bf16<<<gQKV, 256, 0, stream>>>(xb, wqkvb, 1024, 3072, 0, Qw, Kw, Vw, nullptr);

  dim3 gAtt(2048 / 64, 64);
  attn_kernel<<<gAtt, 128, 0, stream>>>(Qw, Kw, Vw, am, attnb);

  dim3 gOut(8192 / BM, 1024 / BN);
  gemm_bf16<<<gOut, 256, 0, stream>>>(attnb, woutb, 1024, 1024, 1,
                                      nullptr, nullptr, nullptr, out);
}
